// PKAttention_80642305950536
// MI455X (gfx1250) — matrix operand order repaired and checked by reading, not compiled, not run on hardware
//
#include <hip/hip_runtime.h>

typedef _Float16 f16;
typedef unsigned short u16;
typedef __attribute__((ext_vector_type(16))) _Float16 v16h;
typedef __attribute__((ext_vector_type(8)))  _Float16 h8;
typedef __attribute__((ext_vector_type(8)))  float    v8f;
typedef __attribute__((ext_vector_type(4)))  unsigned int u32x4;
typedef __attribute__((ext_vector_type(8)))  int i32x8;
typedef __attribute__((ext_vector_type(4)))  int i32x4;

#define DIMX    256
#define HEADS   8
#define NUM_KVX 16384
#define NKEYS   128
#define DKX     128
#define TOPK    16
#define BB      2
#define SSL     1024
#define BS      (BB * SSL)       // 2048 token rows
#define HD      (HEADS * DIMX)   // 2048

union Frag {
  v16h v;
  h8   h[2];
  f16  e[16];
  u16  u[16];
};

static __device__ __forceinline__ v8f wmma_f16(v16h a, v16h b, v8f c) {
  // D = A(16x32 f16) * B(32x16 f16) + C(16x16 f32)
  return __builtin_amdgcn_wmma_f32_16x16x32_f16(false, a, false, b, (short)0, c,
                                                false, false);
}

static __device__ __forceinline__ u16 f2h_bits(float f) {
  union { f16 h; u16 u; } cv;
  cv.h = (f16)f;
  return cv.u;
}

static __device__ __forceinline__ v8f vzero8() {
  v8f z;
#pragma unroll
  for (int i = 0; i < 8; i++) z[i] = 0.f;
  return z;
}

// Tensor Data Mover: DMA a (rows x 256) f16 tile (row stride = HD elems) from
// global memory into LDS.  D# layout per CDNA5 ISA.
// This toolchain exposes the 6-arg builtin form:
//   (uint32x4 g0, int32x8 g1, int32x4 g2, int32x4 g3, int32x8 gx, i32 cpol)
static __device__ __forceinline__ void tdm_load_tile_f16(const u16* gsrc,
                                                         unsigned ldsAddr,
                                                         int rows) {
  unsigned long long ga = (unsigned long long)gsrc;
  u32x4 g0;
  g0[0] = 1u;                                   // count=1, user descriptor
  g0[1] = ldsAddr;                              // lds_addr (bytes)
  g0[2] = (unsigned)ga;                         // global_addr[31:0]
  g0[3] = ((unsigned)(ga >> 32) & 0x01FFFFFFu)  // global_addr[56:32]
          | 0x80000000u;                        // type = 2 ("image")
  i32x8 g1;
  g1[0] = 0x00010000;            // workgroup_mask=0, data_size=1 (2 bytes)
  g1[1] = 0x01000000;            // tensor_dim0 = 256 (bits 79:48, low half here)
  g1[2] = (rows << 16);          // tensor_dim0 hi = 0, tensor_dim1 = rows
  g1[3] = 0x01000000;            // tensor_dim1 hi = 0, tile_dim0 = 256
  g1[4] = rows;                  // tile_dim1 = rows, tile_dim2 = 0
  g1[5] = HD;                    // tensor_dim0_stride = 2048 elems
  g1[6] = 0;                     // stride0 hi, tensor_dim1_stride lo
  g1[7] = 0;
  i32x4 gz;
  gz[0] = 0; gz[1] = 0; gz[2] = 0; gz[3] = 0;   // groups 2/3 unused (2D tensor)
  i32x8 gz8;
#pragma unroll
  for (int i = 0; i < 8; i++) gz8[i] = 0;
  __builtin_amdgcn_tensor_load_to_lds(g0, g1, gz, gz, gz8, 0);
}

// ---------------------------------------------------------------------------
// f32 -> f16 converters (one-time prep so every GEMM runs pure f16 WMMA)
// ---------------------------------------------------------------------------
__global__ __launch_bounds__(256) void conv_f32_f16(const float* __restrict__ s,
                                                    u16* __restrict__ d, int n) {
  int i = blockIdx.x * 256 + threadIdx.x;
  if (i < n) d[i] = f2h_bits(s[i]);
}

// src is K x N row-major f32; dst is N x K row-major f16 (i.e. B^T layout)
__global__ __launch_bounds__(256) void tconv_f32_f16(const float* __restrict__ s,
                                                     u16* __restrict__ d, int K, int N) {
  int i = blockIdx.x * 256 + threadIdx.x;
  if (i >= K * N) return;
  int k = i / N, n = i - k * N;
  d[(long)n * K + k] = f2h_bits(s[i]);
}

// pk_keys (P, NKEYS, H, DK) f32 -> (P, H, NKEYS, DK) f16  (batched B^T layout)
__global__ __launch_bounds__(256) void permute_pk(const float* __restrict__ s,
                                                  u16* __restrict__ d) {
  int i = blockIdx.x * 256 + threadIdx.x;
  if (i >= 2 * NKEYS * HEADS * DKX) return;
  int dd = i & 127;
  int h  = (i >> 7) & 7;
  int n  = (i >> 10) & 127;
  int p  = i >> 17;
  d[(((long)(p * 8 + h)) * NKEYS + n) * DKX + dd] = f2h_bits(s[i]);
}

// ---------------------------------------------------------------------------
// Generic batched WMMA GEMM: C[m,n] = scale * sum_k A[m,k] * BT[n,k]
// One wave computes a 16x64 tile: one A fragment feeds 4 B fragments.
// A frag: lane m=lane&15, k = 8*half + j (j<8) and 16 + 8*half + j
// B frag: lane n=lane&15, k = 8*half + j (j<8) and 16 + 8*half + j
// ---------------------------------------------------------------------------
__global__ __launch_bounds__(32)
void gemm_wmma(const u16* __restrict__ A, int lda, long aBatch,
               const u16* __restrict__ BT, int ldb, long bBatch,
               float* __restrict__ Cf, u16* __restrict__ Ch, int ldc, long cBatch,
               int K, float scale) {
  const int lane = threadIdx.x;
  const int r = lane & 15, half = lane >> 4;
  const int n0 = blockIdx.x * 64, m0 = blockIdx.y * 16;
  const u16* Ap = A + (long)blockIdx.z * aBatch + (long)(m0 + r) * lda;
  const u16* Bp = BT + (long)blockIdx.z * bBatch + (long)(n0 + r) * ldb;
  v8f acc[4];
#pragma unroll
  for (int s4 = 0; s4 < 4; s4++) acc[s4] = vzero8();
  for (int kk = 0; kk < K; kk += 32) {
    Frag a;
    a.h[0] = *(const h8*)(Ap + kk + half * 8);
    a.h[1] = *(const h8*)(Ap + kk + 16 + half * 8);
#pragma unroll
    for (int s4 = 0; s4 < 4; s4++) {
      Frag b;
      const u16* Bs = Bp + (long)s4 * 16 * ldb + kk;
      b.h[0] = *(const h8*)(Bs + half * 8);
      b.h[1] = *(const h8*)(Bs + 16 + half * 8);
      acc[s4] = wmma_f16(a.v, b.v, acc[s4]);
    }
  }
  const long cb = (long)blockIdx.z * cBatch;
#pragma unroll
  for (int s4 = 0; s4 < 4; s4++) {
    if (Ch) {
#pragma unroll
      for (int vv = 0; vv < 8; vv++)
        Ch[cb + (long)(m0 + vv + 8 * half) * ldc + n0 + s4 * 16 + r] =
            f2h_bits(acc[s4][vv] * scale);
    } else {
#pragma unroll
      for (int vv = 0; vv < 8; vv++)
        Cf[cb + (long)(m0 + vv + 8 * half) * ldc + n0 + s4 * 16 + r] =
            acc[s4][vv] * scale;
    }
  }
}

// ---------------------------------------------------------------------------
// Product-key top-k + combine + softmax. One wave per (b,s,h).
// ---------------------------------------------------------------------------
__global__ __launch_bounds__(256)
void pk_topk_kernel(const float* __restrict__ sim, float* __restrict__ wout,
                    int* __restrict__ fiout) {
  const int lane = threadIdx.x & 31;
  const int item = blockIdx.x * 8 + (threadIdx.x >> 5);  // bs*8 + h
  const int bs = item >> 3, h = item & 7;
  const float* base = sim + (long)bs * 2048 + h * NKEYS;  // + p*1024 + key
  float sv[2][4];
#pragma unroll
  for (int p = 0; p < 2; p++)
#pragma unroll
    for (int i = 0; i < 4; i++) sv[p][i] = base[p * 1024 + i * 32 + lane];

  float sel_s[2][16];
  int sel_i[2][16];
#pragma unroll
  for (int p = 0; p < 2; p++) {
    for (int t = 0; t < 16; t++) {
      float bv = -3.4e38f;
      int bi = 0;
#pragma unroll
      for (int i = 0; i < 4; i++) {
        int ki = i * 32 + lane;
        if (sv[p][i] > bv) { bv = sv[p][i]; bi = ki; }
      }
#pragma unroll
      for (int off = 16; off >= 1; off >>= 1) {
        float ov = __shfl_xor(bv, off, 32);
        int oi = __shfl_xor(bi, off, 32);
        if (ov > bv || (ov == bv && oi < bi)) { bv = ov; bi = oi; }
      }
      sel_s[p][t] = bv;
      sel_i[p][t] = bi;
#pragma unroll
      for (int i = 0; i < 4; i++)
        if (i * 32 + lane == bi) sv[p][i] = -3.4e38f;
    }
  }
  // 256 cartesian sums, lane owns combos i*32+lane
  float cv[8];
#pragma unroll
  for (int i = 0; i < 8; i++) {
    int c = i * 32 + lane;
    cv[i] = sel_s[0][c >> 4] + sel_s[1][c & 15];
  }
  float fs[16];
  int fidx[16];
  for (int t = 0; t < 16; t++) {
    float bv = -3.4e38f;
    int bc = 0;
#pragma unroll
    for (int i = 0; i < 8; i++) {
      int c = i * 32 + lane;
      if (cv[i] > bv) { bv = cv[i]; bc = c; }
    }
#pragma unroll
    for (int off = 16; off >= 1; off >>= 1) {
      float ov = __shfl_xor(bv, off, 32);
      int oc = __shfl_xor(bc, off, 32);
      if (ov > bv || (ov == bv && oc < bc)) { bv = ov; bc = oc; }
    }
    fs[t] = bv;
    fidx[t] = sel_i[0][bc >> 4] + sel_i[1][bc & 15] * NKEYS;
#pragma unroll
    for (int i = 0; i < 8; i++)
      if (i * 32 + lane == bc) cv[i] = -3.4e38f;
  }
  float fm = fs[0];
#pragma unroll
  for (int t = 1; t < 16; t++) fm = fmaxf(fm, fs[t]);
  float fsum = 0.f;
#pragma unroll
  for (int t = 0; t < 16; t++) { fs[t] = __expf(fs[t] - fm); fsum += fs[t]; }
#pragma unroll
  for (int t = 0; t < 16; t++)
    if (lane == t) {
      wout[(long)item * 16 + t] = fs[t] / fsum;
      fiout[(long)item * 16 + t] = fidx[t];
    }
}

// ---------------------------------------------------------------------------
// Weighted gather of key/value rows. One wave per (b,s,h); 8 f32 per lane.
// This is the HBM-bound stage (~0.5 GB gathered) -> float4 loads + prefetch.
// ---------------------------------------------------------------------------
__global__ __launch_bounds__(256)
void gather_kv_kernel(const float* __restrict__ w, const int* __restrict__ fi,
                      const float* __restrict__ keyT, const float* __restrict__ valT,
                      u16* __restrict__ kout, u16* __restrict__ vout) {
  const int lane = threadIdx.x & 31;
  const int item = blockIdx.x * 8 + (threadIdx.x >> 5);  // bs*8 + h
  const int bs = item >> 3, h = item & 7;
  const int d0 = lane * 8;
  float ka[8], va[8];
#pragma unroll
  for (int i = 0; i < 8; i++) { ka[i] = 0.f; va[i] = 0.f; }
  long row = ((long)fi[(long)item * 16] + (long)h * NUM_KVX) * DIMX;
  for (int t = 0; t < 16; t++) {
    const float wt = w[(long)item * 16 + t];
    const long rowNext =
        (t < 15) ? ((long)fi[(long)item * 16 + t + 1] + (long)h * NUM_KVX) * DIMX : row;
    __builtin_prefetch(keyT + rowNext + d0, 0, 1);
    __builtin_prefetch(valT + rowNext + d0, 0, 1);
    const float4* kp = (const float4*)(keyT + row + d0);
    const float4* vp = (const float4*)(valT + row + d0);
    float4 k1 = kp[0], k2 = kp[1];
    float4 v1 = vp[0], v2 = vp[1];
    ka[0] += wt * k1.x; ka[1] += wt * k1.y; ka[2] += wt * k1.z; ka[3] += wt * k1.w;
    ka[4] += wt * k2.x; ka[5] += wt * k2.y; ka[6] += wt * k2.z; ka[7] += wt * k2.w;
    va[0] += wt * v1.x; va[1] += wt * v1.y; va[2] += wt * v1.z; va[3] += wt * v1.w;
    va[4] += wt * v2.x; va[5] += wt * v2.y; va[6] += wt * v2.z; va[7] += wt * v2.w;
    row = rowNext;
  }
  h8 hk, hv;
#pragma unroll
  for (int i = 0; i < 8; i++) {
    ((u16*)&hk)[i] = f2h_bits(ka[i]);
    ((u16*)&hv)[i] = f2h_bits(va[i]);
  }
  *(h8*)(kout + (long)bs * HD + h * DIMX + d0) = hk;
  *(h8*)(vout + (long)bs * HD + h * DIMX + d0) = hv;
}

// ---------------------------------------------------------------------------
// Flash attention, causal. 4 waves/WG share K/V chunks (32 keys) staged into
// LDS by the Tensor Data Mover (wave 0 issues TENSOR_LOAD_TO_LDS, waits on
// TENSORcnt, then the WG barrier publishes the tile).  V is then transposed
// LDS->LDS so V^T fragments load as aligned ds_load_b128.
// Wave computes S^T = K * q^T and out^T = V^T * P^T so softmax rows map to
// lane columns (reductions are one shfl_xor(16)).
// ---------------------------------------------------------------------------
__global__ __launch_bounds__(128)
void attn_kernel(const u16* __restrict__ q, const u16* __restrict__ k,
                 const u16* __restrict__ v, u16* __restrict__ ao) {
  __shared__ u16 kt[32 * 256];    // K chunk, row-major (key, dim)   (TDM dest)
  __shared__ u16 vrow[32 * 256];  // V chunk, row-major (key, dim)   (TDM dest)
  __shared__ u16 vt[256 * 32];    // V chunk transposed (dim, key)
  const int tid = threadIdx.x;
  const int wave = tid >> 5, lane = tid & 31;
  const int r = lane & 15, half = lane >> 4;
  const int bh = blockIdx.x;   // b*8 + h
  const int qb4 = blockIdx.y;  // 64-query super block
  const int b = bh >> 3, h = bh & 7;
  const int qBlock = qb4 * 4 + wave;
  const long base = (long)b * SSL * HD + (long)h * DIMX;  // + s*HD + d

  // q fragments, B-operand layout (lane = query column r), q pre-scaled
  Frag qf[8];
  {
    const u16* qrow = q + base + (long)(qBlock * 16 + r) * HD;
#pragma unroll
    for (int kc = 0; kc < 8; kc++) {
      qf[kc].h[0] = *(const h8*)(qrow + kc * 32 + half * 8);
      qf[kc].h[1] = *(const h8*)(qrow + kc * 32 + 16 + half * 8);
    }
  }
  v8f acc[16];
#pragma unroll
  for (int f = 0; f < 16; f++) acc[f] = vzero8();
  float mrow = -3.4e38f, lrow = 0.f;
  const int qi = qBlock * 16 + r;
  const int nChunks = (qb4 + 1) * 2;  // 32 keys per chunk, causal bound for WG

  for (int jc = 0; jc < nChunks; jc++) {
    const int j0 = jc * 32;
    if (wave == 0) {  // TDM: stage K and V 32x256 tiles into LDS
      tdm_load_tile_f16(k + base + (long)j0 * HD,
                        (unsigned)(unsigned long long)(const void*)kt, 32);
      tdm_load_tile_f16(v + base + (long)j0 * HD,
                        (unsigned)(unsigned long long)(const void*)vrow, 32);
      __builtin_amdgcn_s_wait_tensorcnt(0);
    }
    __syncthreads();
    {  // cooperative LDS->LDS transpose of V: 4 threads per key row
      const int row = tid >> 2, part = tid & 3;
#pragma unroll
      for (int c = 0; c < 8; c++) {
        h8 vv8 = *(const h8*)(&vrow[row * 256 + part * 64 + c * 8]);
        const int dbase = part * 64 + c * 8;
#pragma unroll
        for (int e = 0; e < 8; e++) vt[(dbase + e) * 32 + row] = ((const u16*)&vv8)[e];
      }
    }
    __syncthreads();
    if (j0 <= qBlock * 16 + 15) {
      v8f s0 = vzero8(), s1 = vzero8();
#pragma unroll
      for (int kc = 0; kc < 8; kc++) {
        Frag ka;
        const u16* k0p = &kt[r * 256 + kc * 32];
        ka.h[0] = *(const h8*)(k0p + half * 8);
        ka.h[1] = *(const h8*)(k0p + 16 + half * 8);
        s0 = wmma_f16(ka.v, qf[kc].v, s0);  // keys j0..j0+15
        const u16* k1p = &kt[(16 + r) * 256 + kc * 32];
        ka.h[0] = *(const h8*)(k1p + half * 8);
        ka.h[1] = *(const h8*)(k1p + 16 + half * 8);
        s1 = wmma_f16(ka.v, qf[kc].v, s1);  // keys j0+16..j0+31
      }
      // causal mask + online softmax (row == query == lane column)
      float mx = -3.4e38f;
#pragma unroll
      for (int vv = 0; vv < 8; vv++) {
        const int key0 = j0 + vv + 8 * half;
        if (key0 > qi) s0[vv] = -3.4e38f;
        if (key0 + 16 > qi) s1[vv] = -3.4e38f;
        mx = fmaxf(mx, fmaxf(s0[vv], s1[vv]));
      }
      mx = fmaxf(mx, __shfl_xor(mx, 16, 32));
      const float mnew = fmaxf(mrow, mx);
      const float scl = __expf(mrow - mnew);
      float p0[8], p1[8], rs = 0.f;
#pragma unroll
      for (int vv = 0; vv < 8; vv++) {
        p0[vv] = __expf(s0[vv] - mnew);
        p1[vv] = __expf(s1[vv] - mnew);
        rs += p0[vv] + p1[vv];
      }
      rs += __shfl_xor(rs, 16, 32);
      lrow = lrow * scl + rs;
      mrow = mnew;
      // repack P^T (D-frag layout) into f16 B-frag: halves exchange keys
      Frag pf;
#pragma unroll
      for (int vv = 0; vv < 8; vv++) {
        const float o0 = __shfl_xor(p0[vv], 16, 32);
        const float o1 = __shfl_xor(p1[vv], 16, 32);
        pf.e[vv]     = (f16)(half ? o1 : p0[vv]);  // keys 16*half + vv
        pf.e[8 + vv] = (f16)(half ? p1[vv] : o0);  // keys 16*half + 8 + vv
      }
      // out^T += V^T * P^T over 16 dim blocks
#pragma unroll
      for (int f = 0; f < 16; f++) {
#pragma unroll
        for (int vv = 0; vv < 8; vv++) acc[f][vv] *= scl;
        Frag va;
        const u16* vp = &vt[(f * 16 + r) * 32];
        va.h[0] = *(const h8*)(vp + half * 16);
        va.h[1] = *(const h8*)(vp + half * 16 + 8);
        acc[f] = wmma_f16(va.v, pf.v, acc[f]);
      }
    }
    __syncthreads();
  }
  const float inv = 1.f / lrow;
  u16* orow = ao + base + (long)(qBlock * 16 + r) * HD;
#pragma unroll
  for (int f = 0; f < 16; f++)
#pragma unroll
    for (int vv = 0; vv < 8; vv++)
      orow[f * 16 + vv + 8 * half] = f2h_bits(acc[f][vv] * inv);
}

// ---------------------------------------------------------------------------
extern "C" void kernel_launch(void* const* d_in, const int* in_sizes, int n_in,
                              void* d_out, int out_size, void* d_ws, size_t ws_size,
                              hipStream_t stream) {
  const float* x      = (const float*)d_in[0];
  const float* wq_pk  = (const float*)d_in[1];
  const float* pkkeys = (const float*)d_in[2];
  const float* wq     = (const float*)d_in[3];
  const float* keyT   = (const float*)d_in[4];
  const float* valT   = (const float*)d_in[5];
  const float* wo     = (const float*)d_in[6];
  float* out = (float*)d_out;

  char* p = (char*)d_ws;
  size_t off = 0;
  auto take = [&](size_t bytes) -> void* {
    void* r = p + off;
    off += (bytes + 255) & ~(size_t)255;
    return r;
  };
  u16*   x_h    = (u16*)take((size_t)BS * DIMX * 2);  // 2048x256
  u16*   wqpk_t = (u16*)take((size_t)HD * DIMX * 2);  // 2048x256 (N x K)
  u16*   wq_t   = (u16*)take((size_t)HD * DIMX * 2);  // 2048x256
  u16*   wo_t   = (u16*)take((size_t)DIMX * HD * 2);  // 256x2048
  u16*   pk_h   = (u16*)take((size_t)2 * HEADS * NKEYS * DKX * 2);
  u16*   qpk_h  = (u16*)take((size_t)BS * HD * 2);    // 2048x2048
  float* simb   = (float*)take((size_t)BS * 2048 * 4);  // (bs, p, h, key)
  float* wbuf   = (float*)take((size_t)BS * HEADS * TOPK * 4);
  int*   fibuf  = (int*)take((size_t)BS * HEADS * TOPK * 4);
  u16*   q_h    = (u16*)take((size_t)BS * HD * 2);
  u16*   k_h    = (u16*)take((size_t)BS * HD * 2);
  u16*   v_h    = (u16*)take((size_t)BS * HD * 2);
  u16*   ao_h   = (u16*)take((size_t)BS * HD * 2);
  (void)ws_size; (void)in_sizes; (void)n_in; (void)out_size;

  // 1) one-time f16 conversion / weight transposition
  conv_f32_f16<<<(BS * DIMX + 255) / 256, 256, 0, stream>>>(x, x_h, BS * DIMX);
  tconv_f32_f16<<<(DIMX * HD + 255) / 256, 256, 0, stream>>>(wq_pk, wqpk_t, DIMX, HD);
  tconv_f32_f16<<<(DIMX * HD + 255) / 256, 256, 0, stream>>>(wq, wq_t, DIMX, HD);
  tconv_f32_f16<<<(HD * DIMX + 255) / 256, 256, 0, stream>>>(wo, wo_t, HD, DIMX);
  permute_pk<<<(2 * NKEYS * HEADS * DKX + 255) / 256, 256, 0, stream>>>(pkkeys, pk_h);

  // 2) q_pk = x @ wq_pk  (2048x2048, K=256) -> f16
  gemm_wmma<<<dim3(HD / 64, BS / 16, 1), 32, 0, stream>>>(
      x_h, DIMX, 0, wqpk_t, DIMX, 0, nullptr, qpk_h, HD, 0, DIMX, 1.f);

  // 3) sim[bs, p, h, key] : batched over z = p*8+h, K = 128
  gemm_wmma<<<dim3(NKEYS / 64, BS / 16, 16), 32, 0, stream>>>(
      qpk_h, HD, DKX, pk_h, DKX, (long)NKEYS * DKX,
      simb, nullptr, 2048, NKEYS, DKX, 1.f);

  // 4) per-(b,s,h) product-key top-k + softmax
  pk_topk_kernel<<<BS * HEADS / 8, 256, 0, stream>>>(simb, wbuf, fibuf);

  // 5) q = (x @ wq) * DIM^-0.5 -> f16 (layout bs, h*256+d)
  gemm_wmma<<<dim3(HD / 64, BS / 16, 1), 32, 0, stream>>>(
      x_h, DIMX, 0, wq_t, DIMX, 0, nullptr, q_h, HD, 0, DIMX, 0.0625f);

  // 6) weighted gather of k/v rows (HBM-bound stage)
  gather_kv_kernel<<<BS * HEADS / 8, 256, 0, stream>>>(wbuf, fibuf, keyT, valT, k_h, v_h);

  // 7) causal flash attention (TDM-staged K/V) -> ao (bs, h*256+d) f16
  attn_kernel<<<dim3(BB * HEADS, SSL / 64), 128, 0, stream>>>(q_h, k_h, v_h, ao_h);

  // 8) out = ao @ wo  (2048x256, K=2048) -> f32 d_out
  gemm_wmma<<<dim3(DIMX / 64, BS / 16, 1), 32, 0, stream>>>(
      ao_h, HD, 0, wo_t, HD, 0, out, nullptr, DIMX, 0, HD, 1.f);
}
